// Encoder_86423331930253
// MI455X (gfx1250) — matrix operand order repaired and checked by reading, not compiled, not run on hardware
//
#include <hip/hip_runtime.h>
#include <hip/hip_fp16.h>
#include <math.h>
#include <stdint.h>

typedef __attribute__((ext_vector_type(16))) _Float16 v16h;
typedef __attribute__((ext_vector_type(8)))  _Float16 v8h;
typedef __attribute__((ext_vector_type(8)))  float    v8f;

#define LNUM   6
#define HNUM   16
#define DMODEL 1024
#define DKH    64
#define DFF    4096
#define BB     2
#define SS     1024
#define MROWS  (BB*SS)

static __device__ inline v16h cat8(v8h lo, v8h hic) {
  v16h r;
  #pragma unroll
  for (int i = 0; i < 8; ++i) { r[i] = lo[i]; r[i + 8] = hic[i]; }
  return r;
}

// CDNA5 async LDS DMA: 16B per active lane, tracked with ASYNCcnt.
// LDS dest operand is the wave-relative LDS byte offset == low 32 bits of the
// generic pointer to __shared__ (ISA 10.2).
static __device__ inline void async_copy_b128(const _Float16* g, _Float16* l) {
  unsigned lo = (unsigned)(uintptr_t)l;
  asm volatile("global_load_async_to_lds_b128 %0, %1, off"
               :: "v"(lo), "v"(g) : "memory");
}
static __device__ inline void wait_async0() {
  asm volatile("s_wait_asynccnt 0" ::: "memory");
}

// ---------------------------------------------------------------------------
// Positional encoding: x = src + PE ; PE[s,d] = (d even) ? sin(s) : cos(s)
// ---------------------------------------------------------------------------
__global__ __launch_bounds__(256) void pos_encode_kernel(
    const float* __restrict__ src, float* __restrict__ xf,
    _Float16* __restrict__ xh) {
  int idx = blockIdx.x * 256 + threadIdx.x;
  int d = idx & (DMODEL - 1);
  int srow = (idx / DMODEL) & (SS - 1);
  float pos = (float)srow;
  float pe = (d & 1) ? cosf(pos) : sinf(pos);
  float v = src[idx] + pe;
  xf[idx] = v;
  xh[idx] = (_Float16)v;
}

// ---------------------------------------------------------------------------
// Weight convert + transpose: W_f32[K][N] (optionally head-blocked (H,K,64))
// -> Wt_f16[N][K].  Done once per launch.
// ---------------------------------------------------------------------------
__global__ __launch_bounds__(256) void transpose_w_kernel(
    const float* __restrict__ W, _Float16* __restrict__ Wt,
    int K, int N, int head_blocked) {
  __shared__ float T[32][33];
  int k0 = blockIdx.x * 32, n0 = blockIdx.y * 32;
  int tid = threadIdx.x;
  int tx = tid & 31, tyb = tid >> 5;
  const float* srcp;
  size_t stride;
  if (head_blocked) { srcp = W + (size_t)(n0 >> 6) * K * 64 + (n0 & 63) + tx; stride = 64; }
  else              { srcp = W + (size_t)n0 + tx;                             stride = (size_t)N; }
  #pragma unroll
  for (int j = 0; j < 4; ++j) {
    int k = tyb + 8 * j;
    T[k][tx] = srcp[(size_t)(k0 + k) * stride];
  }
  __syncthreads();
  #pragma unroll
  for (int j = 0; j < 4; ++j) {
    int n = tyb + 8 * j;
    Wt[(size_t)(n0 + n) * K + k0 + tx] = (_Float16)T[tx][n];
  }
}

// ---------------------------------------------------------------------------
// WMMA GEMM:  C[M,N] = A_f16[M,K] * Wt_f16[N][K]^T (+bias) (+relu)
// Tile 128(M) x 64(N), BK=64, 128 threads (4 waves); each wave owns a 32x64
// strip -> 2 A-frags x 4 B-frags = 16 WMMAs per stage.  Double-buffered LDS
// filled by global_load_async_to_lds_b128; stage i+1 DMA overlaps stage i math.
// vtr: store output transposed+packed as v_t[b][h][dk][s].
// ---------------------------------------------------------------------------
#define GP 72   // padded LDS row stride in halves (144B, 16B-aligned)
__global__ __launch_bounds__(128) void gemm_kernel(
    const _Float16* __restrict__ A, const _Float16* __restrict__ Bt,
    const float* __restrict__ bias, _Float16* __restrict__ outh,
    float* __restrict__ outf, int N, int K, int relu, int vtr) {
  __shared__ __align__(16) _Float16 As[2 * 128 * GP];
  __shared__ __align__(16) _Float16 Bs[2 * 64 * GP];
  int n0 = blockIdx.x * 64, m0 = blockIdx.y * 128;
  int tid = threadIdx.x, lane = tid & 31, wave = tid >> 5;
  int hi = lane >> 4, ln = lane & 15;
  int rb = tid >> 3, c = (tid & 7) * 8;   // 8-half chunk base: row rb(+16j), col c

  const _Float16* Ap = A + (size_t)(m0 + rb) * K + c;
  const _Float16* Bp = Bt + (size_t)(n0 + rb) * K + c;
  _Float16* lA = &As[rb * GP + c];
  _Float16* lB = &Bs[rb * GP + c];

  // prologue: DMA stage 0 into buffer 0 (A: 8 chunks, B: 4 chunks per thread)
  #pragma unroll
  for (int j = 0; j < 8; ++j) async_copy_b128(Ap + (size_t)16 * j * K, lA + 16 * j * GP);
  #pragma unroll
  for (int j = 0; j < 4; ++j) async_copy_b128(Bp + (size_t)16 * j * K, lB + 16 * j * GP);

  v8f acc[2][4] = {};
  int ns = K >> 6;
  for (int it = 0; it < ns; ++it) {
    wait_async0();
    __syncthreads();                      // buf[cur] ready; buf[nxt] free
    int curA = (it & 1) * (128 * GP), nxtA = ((it + 1) & 1) * (128 * GP);
    int curB = (it & 1) * (64 * GP),  nxtB = ((it + 1) & 1) * (64 * GP);
    if (it + 1 < ns) {
      size_t koff = (size_t)(it + 1) << 6;
      #pragma unroll
      for (int j = 0; j < 8; ++j)
        async_copy_b128(Ap + (size_t)16 * j * K + koff, lA + nxtA + 16 * j * GP);
      #pragma unroll
      for (int j = 0; j < 4; ++j)
        async_copy_b128(Bp + (size_t)16 * j * K + koff, lB + nxtB + 16 * j * GP);
    }
    #pragma unroll
    for (int ks = 0; ks < 2; ++ks) {
      int ko = ks * 32;
      v16h af[2];
      #pragma unroll
      for (int a = 0; a < 2; ++a) {
        int mr = curA + (wave * 32 + a * 16 + ln) * GP + ko;
        af[a] = cat8(*(const v8h*)&As[mr + 8 * hi], *(const v8h*)&As[mr + 16 + 8 * hi]);
      }
      v16h bf[4];
      #pragma unroll
      for (int g = 0; g < 4; ++g) {
        int nr = curB + (g * 16 + ln) * GP + ko;
        bf[g] = cat8(*(const v8h*)&Bs[nr + 8 * hi], *(const v8h*)&Bs[nr + 16 + 8 * hi]);
      }
      #pragma unroll
      for (int a = 0; a < 2; ++a)
        #pragma unroll
        for (int g = 0; g < 4; ++g)
          acc[a][g] = __builtin_amdgcn_wmma_f32_16x16x32_f16(
              false, af[a], false, bf[g], (short)0, acc[a][g], false, false);
    }
  }
  // epilogue
  #pragma unroll
  for (int a = 0; a < 2; ++a) {
    #pragma unroll
    for (int g = 0; g < 4; ++g) {
      int col = n0 + g * 16 + ln;
      float bvv = bias ? bias[col] : 0.0f;
      if (vtr) {
        int hh = col >> 6, dk = col & 63;
        int row0 = m0 + wave * 32 + a * 16 + 8 * hi;   // 8 consecutive rows (s)
        int bidx = row0 >> 10, s0 = row0 & (SS - 1);
        v8h pk;
        #pragma unroll
        for (int vv = 0; vv < 8; ++vv) pk[vv] = (_Float16)(acc[a][g][vv] + bvv);
        *(v8h*)&outh[(((size_t)bidx * HNUM + hh) * DKH + dk) * SS + s0] = pk;
      } else {
        #pragma unroll
        for (int vv = 0; vv < 8; ++vv) {
          int row = m0 + wave * 32 + a * 16 + vv + 8 * hi;
          float val = acc[a][g][vv] + bvv;
          if (relu) val = fmaxf(val, 0.0f);
          if (outf) outf[(size_t)row * N + col] = val;
          if (outh) outh[(size_t)row * N + col] = (_Float16)val;
        }
      }
    }
  }
}

// ---------------------------------------------------------------------------
// Flash attention per (b, h, 64-row Q tile). f16 WMMA, f32 online softmax.
// Q/K/V tiles arrive in LDS via async DMA; V is pre-transposed globally.
// ---------------------------------------------------------------------------
#define AP 72   // padded LDS row stride in halves (144B, 16B-aligned)
__global__ __launch_bounds__(128) void attn_kernel(
    const _Float16* __restrict__ Q, const _Float16* __restrict__ Kt,
    const _Float16* __restrict__ Vt, _Float16* __restrict__ O) {
  __shared__ __align__(16) _Float16 Qs[64 * AP];
  __shared__ __align__(16) _Float16 Ks[64 * AP];
  __shared__ __align__(16) _Float16 Vs[64 * AP];   // Vs[dk][key]
  __shared__ __align__(16) _Float16 Ps[4 * 16 * AP];
  int mt = blockIdx.x, h = blockIdx.y, b = blockIdx.z;
  int tid = threadIdx.x, lane = tid & 31, wave = tid >> 5;
  int hi = lane >> 4, ln = lane & 15;
  const size_t ld = (size_t)HNUM * DKH;
  const _Float16* qb = Q + ((size_t)b * SS + (size_t)mt * 64) * ld + (size_t)h * DKH;
  const _Float16* kb = Kt + (size_t)b * SS * ld + (size_t)h * DKH;
  const _Float16* vtb = Vt + ((size_t)b * HNUM + h) * DKH * SS;  // [64][SS]

  #pragma unroll
  for (int j = 0; j < 4; ++j) {
    int i = tid + 128 * j; int r = i >> 3, cc = (i & 7) * 8;
    async_copy_b128(&qb[(size_t)r * ld + cc], &Qs[r * AP + cc]);
  }
  wait_async0();
  __syncthreads();
  int mrow = wave * 16 + ln;
  v16h qa0 = cat8(*(const v8h*)&Qs[mrow * AP + 8 * hi],
                  *(const v8h*)&Qs[mrow * AP + 16 + 8 * hi]);
  v16h qa1 = cat8(*(const v8h*)&Qs[mrow * AP + 32 + 8 * hi],
                  *(const v8h*)&Qs[mrow * AP + 48 + 8 * hi]);
  v8f oacc[4] = {};
  float mrun[8], lrun[8];
  #pragma unroll
  for (int vv = 0; vv < 8; ++vv) { mrun[vv] = -1e30f; lrun[vv] = 0.0f; }
  const float scale = 0.125f;   // 1/sqrt(64)

  for (int kt = 0; kt < SS; kt += 64) {
    __syncthreads();                      // previous tile's readers done
    #pragma unroll
    for (int j = 0; j < 4; ++j) {
      int i = tid + 128 * j; int r = i >> 3, cc = (i & 7) * 8;
      async_copy_b128(&kb[(size_t)(kt + r) * ld + cc], &Ks[r * AP + cc]);
    }
    #pragma unroll
    for (int j = 0; j < 4; ++j) {
      int i = tid + 128 * j; int r = i >> 3, cc = (i & 7) * 8;
      async_copy_b128(&vtb[(size_t)r * SS + kt + cc], &Vs[r * AP + cc]);
    }
    wait_async0();
    __syncthreads();

    v8f sc[4];
    #pragma unroll
    for (int g = 0; g < 4; ++g) {
      int nr = (g * 16 + ln) * AP;
      v16h b0 = cat8(*(const v8h*)&Ks[nr + 8 * hi], *(const v8h*)&Ks[nr + 16 + 8 * hi]);
      v16h b1 = cat8(*(const v8h*)&Ks[nr + 32 + 8 * hi], *(const v8h*)&Ks[nr + 48 + 8 * hi]);
      v8f z = {};
      z = __builtin_amdgcn_wmma_f32_16x16x32_f16(false, qa0, false, b0, (short)0, z, false, false);
      sc[g] = __builtin_amdgcn_wmma_f32_16x16x32_f16(false, qa1, false, b1, (short)0, z, false, false);
    }
    float mnew[8], corr[8];
    #pragma unroll
    for (int vv = 0; vv < 8; ++vv) {
      float t0 = fmaxf(fmaxf(sc[0][vv], sc[1][vv]), fmaxf(sc[2][vv], sc[3][vv])) * scale;
      #pragma unroll
      for (int off = 8; off >= 1; off >>= 1) t0 = fmaxf(t0, __shfl_xor(t0, off, 32));
      mnew[vv] = fmaxf(mrun[vv], t0);
      corr[vv] = __expf(mrun[vv] - mnew[vv]);
    }
    float rs[8] = {0, 0, 0, 0, 0, 0, 0, 0};
    #pragma unroll
    for (int g = 0; g < 4; ++g) {
      #pragma unroll
      for (int vv = 0; vv < 8; ++vv) {
        float p = __expf(sc[g][vv] * scale - mnew[vv]);
        rs[vv] += p;
        Ps[wave * (16 * AP) + (vv + 8 * hi) * AP + g * 16 + ln] = (_Float16)p;
      }
    }
    #pragma unroll
    for (int vv = 0; vv < 8; ++vv) {
      float t = rs[vv];
      #pragma unroll
      for (int off = 8; off >= 1; off >>= 1) t += __shfl_xor(t, off, 32);
      lrun[vv] = lrun[vv] * corr[vv] + t;
      mrun[vv] = mnew[vv];
    }
    #pragma unroll
    for (int g = 0; g < 4; ++g)
      #pragma unroll
      for (int vv = 0; vv < 8; ++vv) oacc[g][vv] *= corr[vv];

    int pb = wave * (16 * AP) + ln * AP;
    v16h pa0 = cat8(*(const v8h*)&Ps[pb + 8 * hi], *(const v8h*)&Ps[pb + 16 + 8 * hi]);
    v16h pa1 = cat8(*(const v8h*)&Ps[pb + 32 + 8 * hi], *(const v8h*)&Ps[pb + 48 + 8 * hi]);
    #pragma unroll
    for (int g = 0; g < 4; ++g) {
      int nr = (g * 16 + ln) * AP;
      v16h vb0 = cat8(*(const v8h*)&Vs[nr + 8 * hi], *(const v8h*)&Vs[nr + 16 + 8 * hi]);
      oacc[g] = __builtin_amdgcn_wmma_f32_16x16x32_f16(false, pa0, false, vb0, (short)0, oacc[g], false, false);
      v16h vb1 = cat8(*(const v8h*)&Vs[nr + 32 + 8 * hi], *(const v8h*)&Vs[nr + 48 + 8 * hi]);
      oacc[g] = __builtin_amdgcn_wmma_f32_16x16x32_f16(false, pa1, false, vb1, (short)0, oacc[g], false, false);
    }
  }
  #pragma unroll
  for (int g = 0; g < 4; ++g) {
    #pragma unroll
    for (int vv = 0; vv < 8; ++vv) {
      size_t row = (size_t)b * SS + (size_t)mt * 64 + wave * 16 + vv + 8 * hi;
      size_t col = (size_t)h * DKH + g * 16 + ln;
      O[row * ld + col] = (_Float16)(oacc[g][vv] / lrun[vv]);
    }
  }
}

// ---------------------------------------------------------------------------
// Fused residual add + LayerNorm. One block per row. Writes f32 + f16 copy.
// ---------------------------------------------------------------------------
__global__ __launch_bounds__(256) void add_ln_kernel(
    const float* __restrict__ x, const float* __restrict__ r,
    const float* __restrict__ gam, const float* __restrict__ bet,
    float* __restrict__ xout, _Float16* __restrict__ hout) {
  __shared__ float red[256];
  int row = blockIdx.x, tid = threadIdx.x;
  const float* xr = x + (size_t)row * DMODEL;
  const float* rr = r + (size_t)row * DMODEL;
  float v[4]; float s = 0.0f;
  #pragma unroll
  for (int i = 0; i < 4; ++i) { v[i] = xr[tid + i * 256] + rr[tid + i * 256]; s += v[i]; }
  red[tid] = s; __syncthreads();
  for (int o = 128; o > 0; o >>= 1) { if (tid < o) red[tid] += red[tid + o]; __syncthreads(); }
  float mu = red[0] * (1.0f / DMODEL);
  __syncthreads();
  float sq = 0.0f;
  #pragma unroll
  for (int i = 0; i < 4; ++i) { float d = v[i] - mu; sq += d * d; }
  red[tid] = sq; __syncthreads();
  for (int o = 128; o > 0; o >>= 1) { if (tid < o) red[tid] += red[tid + o]; __syncthreads(); }
  float rstd = rsqrtf(red[0] * (1.0f / DMODEL) + 1e-5f);
  #pragma unroll
  for (int i = 0; i < 4; ++i) {
    int cix = tid + i * 256;
    float o = (v[i] - mu) * rstd * gam[cix] + bet[cix];
    xout[(size_t)row * DMODEL + cix] = o;
    hout[(size_t)row * DMODEL + cix] = (_Float16)o;
  }
}

// ---------------------------------------------------------------------------
extern "C" void kernel_launch(void* const* d_in, const int* in_sizes, int n_in,
                              void* d_out, int out_size, void* d_ws, size_t ws_size,
                              hipStream_t stream) {
  (void)in_sizes; (void)n_in; (void)out_size; (void)ws_size;
  const float* src = (const float*)d_in[0];
  const float* Wq  = (const float*)d_in[1];
  const float* bq  = (const float*)d_in[2];
  const float* Wk  = (const float*)d_in[3];
  const float* bk  = (const float*)d_in[4];
  const float* Wv  = (const float*)d_in[5];
  const float* bv  = (const float*)d_in[6];
  const float* Wo  = (const float*)d_in[7];
  const float* bo  = (const float*)d_in[8];
  const float* g1  = (const float*)d_in[9];
  const float* be1 = (const float*)d_in[10];
  const float* W1  = (const float*)d_in[11];
  const float* b1  = (const float*)d_in[12];
  const float* W2  = (const float*)d_in[13];
  const float* b2  = (const float*)d_in[14];
  const float* g2  = (const float*)d_in[15];
  const float* be2 = (const float*)d_in[16];

  char* ws = (char*)d_ws;
  size_t off = 0;
  auto carve = [&](size_t bytes) -> void* {
    void* p = ws + off; off += bytes; off = (off + 255) & ~(size_t)255; return p;
  };
  float*    x_f   = (float*)   carve((size_t)MROWS * DMODEL * 4);
  _Float16* x_h   = (_Float16*)carve((size_t)MROWS * DMODEL * 2);
  _Float16* q_h   = (_Float16*)carve((size_t)MROWS * DMODEL * 2);
  _Float16* k_h   = (_Float16*)carve((size_t)MROWS * DMODEL * 2);
  _Float16* v_t   = (_Float16*)carve((size_t)MROWS * DMODEL * 2);  // [B][H][64][S]
  _Float16* a_h   = (_Float16*)carve((size_t)MROWS * DMODEL * 2);
  float*    tmp_f = (float*)   carve((size_t)MROWS * DMODEL * 4);
  _Float16* h1_h  = (_Float16*)carve((size_t)MROWS * DFF * 2);
  _Float16* wt_q  = (_Float16*)carve((size_t)DMODEL * DMODEL * 2);
  _Float16* wt_k  = (_Float16*)carve((size_t)DMODEL * DMODEL * 2);
  _Float16* wt_v  = (_Float16*)carve((size_t)DMODEL * DMODEL * 2);
  _Float16* wt_o  = (_Float16*)carve((size_t)DMODEL * DMODEL * 2);
  _Float16* wt_1  = (_Float16*)carve((size_t)DMODEL * DFF * 2);
  _Float16* wt_2  = (_Float16*)carve((size_t)DFF * DMODEL * 2);

  pos_encode_kernel<<<dim3((MROWS * DMODEL) / 256), dim3(256), 0, stream>>>(src, x_f, x_h);

  dim3 gD(DMODEL / 64, MROWS / 128);
  dim3 gF(DFF / 64, MROWS / 128);
  dim3 tDD(DMODEL / 32, DMODEL / 32);
  dim3 tDF(DMODEL / 32, DFF / 32);
  dim3 tFD(DFF / 32, DMODEL / 32);
  for (int l = 0; l < LNUM; ++l) {
    const float* Wq_l = Wq + (size_t)l * HNUM * DMODEL * DKH;
    const float* Wk_l = Wk + (size_t)l * HNUM * DMODEL * DKH;
    const float* Wv_l = Wv + (size_t)l * HNUM * DMODEL * DKH;
    const float* Wo_l = Wo + (size_t)l * DMODEL * DMODEL;
    const float* W1_l = W1 + (size_t)l * DMODEL * DFF;
    const float* W2_l = W2 + (size_t)l * DFF * DMODEL;

    transpose_w_kernel<<<tDD, 256, 0, stream>>>(Wq_l, wt_q, DMODEL, DMODEL, 1);
    transpose_w_kernel<<<tDD, 256, 0, stream>>>(Wk_l, wt_k, DMODEL, DMODEL, 1);
    transpose_w_kernel<<<tDD, 256, 0, stream>>>(Wv_l, wt_v, DMODEL, DMODEL, 1);
    transpose_w_kernel<<<tDD, 256, 0, stream>>>(Wo_l, wt_o, DMODEL, DMODEL, 0);
    transpose_w_kernel<<<tDF, 256, 0, stream>>>(W1_l, wt_1, DMODEL, DFF, 0);
    transpose_w_kernel<<<tFD, 256, 0, stream>>>(W2_l, wt_2, DFF, DMODEL, 0);

    gemm_kernel<<<gD, 128, 0, stream>>>(x_h, wt_q, bq + (size_t)l * DMODEL, q_h, nullptr,
                                        DMODEL, DMODEL, 0, 0);
    gemm_kernel<<<gD, 128, 0, stream>>>(x_h, wt_k, bk + (size_t)l * DMODEL, k_h, nullptr,
                                        DMODEL, DMODEL, 0, 0);
    gemm_kernel<<<gD, 128, 0, stream>>>(x_h, wt_v, bv + (size_t)l * DMODEL, v_t, nullptr,
                                        DMODEL, DMODEL, 0, 1);
    attn_kernel<<<dim3(SS / 64, HNUM, BB), 128, 0, stream>>>(q_h, k_h, v_t, a_h);
    gemm_kernel<<<gD, 128, 0, stream>>>(a_h, wt_o, bo + (size_t)l * DMODEL, nullptr, tmp_f,
                                        DMODEL, DMODEL, 0, 0);
    add_ln_kernel<<<dim3(MROWS), 256, 0, stream>>>(x_f, tmp_f, g1 + (size_t)l * DMODEL,
                                                   be1 + (size_t)l * DMODEL, x_f, x_h);
    gemm_kernel<<<gF, 128, 0, stream>>>(x_h, wt_1, b1 + (size_t)l * DFF, h1_h, nullptr,
                                        DFF, DMODEL, 1, 0);
    gemm_kernel<<<gD, 128, 0, stream>>>(h1_h, wt_2, b2 + (size_t)l * DMODEL, nullptr, tmp_f,
                                        DMODEL, DFF, 0, 0);
    float* xdst = (l == LNUM - 1) ? (float*)d_out : x_f;
    add_ln_kernel<<<dim3(MROWS), 256, 0, stream>>>(x_f, tmp_f, g2 + (size_t)l * DMODEL,
                                                   be2 + (size_t)l * DMODEL, xdst, x_h);
  }
}
